// HiCHAN_62466004353061
// MI455X (gfx1250) — matrix operand order repaired and checked by reading, not compiled, not run on hardware
//
#include <hip/hip_runtime.h>

// ---------------------------------------------------------------------------
// Shapes (fixed by the reference): Bi=Bt=128, K=36 (padded to 48), L=64, D=1024
// ---------------------------------------------------------------------------
#define BI   128
#define BT   128
#define KDIM 36
#define KPAD 48
#define LDIM 64
#define DDIM 1024
#define LAMB_INV 20.0f   // 1/0.05

typedef __attribute__((ext_vector_type(16))) __bf16        v16bf;
typedef __attribute__((ext_vector_type(8)))  float         v8f;
typedef __attribute__((ext_vector_type(4)))  unsigned int  u32x4;

union Frag {
    u32x4 q[2];   // two 16-byte chunks per lane
    v16bf v;      // operand view for v_wmma_f32_16x16x32_bf16
};

__device__ __forceinline__ unsigned short f2bf_rn(float f) {
    unsigned u = __float_as_uint(f);
    u += 0x7FFFu + ((u >> 16) & 1u);          // round-to-nearest-even
    return (unsigned short)(u >> 16);
}
__device__ __forceinline__ float bf2f(unsigned short h) {
    return __uint_as_float(((unsigned)h) << 16);
}

__device__ __forceinline__ v8f wmma_bf16(v16bf a, v16bf b, v8f c) {
    // (neg_a, A, neg_b, B, c_mod, C, reuse_a, reuse_b)
    return __builtin_amdgcn_wmma_f32_16x16x32_bf16(false, a, false, b,
                                                   (short)0, c, false, false);
}

// ---------------------------------------------------------------------------
// Kernel 1: x = v + 1e-6; x /= ||x||; emit bf16 hi + bf16 lo (error split).
// One 128-thread block per output row. Image rows are padded 36 -> 48 (zeros).
// ---------------------------------------------------------------------------
__global__ __launch_bounds__(128) void normalize_split(
    const float* __restrict__ imgs, const float* __restrict__ caps,
    unsigned short* __restrict__ Ah, unsigned short* __restrict__ Al,
    unsigned short* __restrict__ Bh, unsigned short* __restrict__ Bl)
{
    __shared__ float sm[128];
    const int row = blockIdx.x;
    const int tid = threadIdx.x;

    const float* src = nullptr;
    unsigned short *dh, *dl;
    bool zero = false;

    if (row < BI * KPAD) {                      // image region (padded)
        const int i = row / KPAD, k = row % KPAD;
        dh = Ah + (size_t)row * DDIM;
        dl = Al + (size_t)row * DDIM;
        if (k >= KDIM) zero = true;
        else           src = imgs + ((size_t)i * KDIM + k) * DDIM;
    } else {                                    // caption region (exactly 64 rows)
        const int rr = row - BI * KPAD;
        dh = Bh + (size_t)rr * DDIM;
        dl = Bl + (size_t)rr * DDIM;
        src = caps + (size_t)rr * DDIM;
    }

    if (zero) {                                 // uniform per block: no barrier hazard
        for (int j = tid; j < DDIM; j += 128) { dh[j] = 0; dl[j] = 0; }
        return;
    }

    float x[8];
    float ss = 0.f;
#pragma unroll
    for (int j = 0; j < 8; ++j) {
        x[j] = src[tid + j * 128] + 1e-6f;
        ss += x[j] * x[j];
    }
    sm[tid] = ss;
    __syncthreads();
    for (int off = 64; off > 0; off >>= 1) {
        if (tid < off) sm[tid] += sm[tid + off];
        __syncthreads();
    }
    const float rn = 1.0f / sqrtf(sm[0]);

#pragma unroll
    for (int j = 0; j < 8; ++j) {
        const float n  = x[j] * rn;
        const unsigned short hs = f2bf_rn(n);
        const float lo = n - bf2f(hs);
        dh[tid + j * 128] = hs;
        dl[tid + j * 128] = f2bf_rn(lo);
    }
}

// ---------------------------------------------------------------------------
// Kernel 2: ONE wave32 per (i,t) pair; the wave owns the full 48x64 block
// (3 M-tiles x 4 N-tiles, 12 f32 accumulators). Split-bf16 GEMM:
//    acc += Ahi*Bhi + Alo*Bhi + Ahi*Blo    (36 WMMAs per 32-wide k-step,
//                                           28 b128 loads -> 0.78 loads/WMMA)
// Then fused masked softmax (both axes) + uniform pooling -> sims[i,t].
// Single-wave workgroup: LDS ops are in-order, barriers degenerate to S_NOP.
// ---------------------------------------------------------------------------
__global__ __launch_bounds__(32) void cam_block(
    const unsigned short* __restrict__ Ah, const unsigned short* __restrict__ Al,
    const unsigned short* __restrict__ Bh, const unsigned short* __restrict__ Bl,
    const int* __restrict__ img_lens, const int* __restrict__ cap_lens,
    float* __restrict__ out)
{
    const int t = blockIdx.x;
    const int i = blockIdx.y;

    const int lane = threadIdx.x & 31;
    const int r    = lane & 15;
    const int half = lane >> 4;

    // A bf16 16x32 layout: lane holds row r; chunks at K = half*8 and half*8+16
    const size_t aRow = ((size_t)i * KPAD + r) * DDIM + (size_t)(half * 8);
    const unsigned short* pAh = Ah + aRow;
    const unsigned short* pAl = Al + aRow;
    // B bf16 32x16 layout: lane holds col r; chunks at K = half*8 and half*8+16
    const size_t bRow = ((size_t)t * LDIM + r) * DDIM + (size_t)(half * 8);
    const unsigned short* pBh = Bh + bRow;
    const unsigned short* pBl = Bl + bRow;

    v8f acc[3][4];
#pragma unroll
    for (int m = 0; m < 3; ++m)
#pragma unroll
        for (int n = 0; n < 4; ++n)
            acc[m][n] = (v8f){};

    for (int k = 0; k < DDIM; k += 32) {
        Frag ah[3], al[3], bh[4], bl[4];
#pragma unroll
        for (int m = 0; m < 3; ++m) {
            ah[m].q[0] = *(const u32x4*)(pAh + m * 16 * DDIM + k);
            ah[m].q[1] = *(const u32x4*)(pAh + m * 16 * DDIM + k + 16);
            al[m].q[0] = *(const u32x4*)(pAl + m * 16 * DDIM + k);
            al[m].q[1] = *(const u32x4*)(pAl + m * 16 * DDIM + k + 16);
        }
#pragma unroll
        for (int n = 0; n < 4; ++n) {
            bh[n].q[0] = *(const u32x4*)(pBh + n * 16 * DDIM + k);
            bh[n].q[1] = *(const u32x4*)(pBh + n * 16 * DDIM + k + 16);
            bl[n].q[0] = *(const u32x4*)(pBl + n * 16 * DDIM + k);
            bl[n].q[1] = *(const u32x4*)(pBl + n * 16 * DDIM + k + 16);
        }
        // Three 12-wide sweeps: 12 independent acc chains, no back-to-back
        // RAW on any accumulator (hazard NOPs never needed between neighbors).
#pragma unroll
        for (int m = 0; m < 3; ++m)
#pragma unroll
            for (int n = 0; n < 4; ++n)
                acc[m][n] = wmma_bf16(ah[m].v, bh[n].v, acc[m][n]);
#pragma unroll
        for (int m = 0; m < 3; ++m)
#pragma unroll
            for (int n = 0; n < 4; ++n)
                acc[m][n] = wmma_bf16(al[m].v, bh[n].v, acc[m][n]);
#pragma unroll
        for (int m = 0; m < 3; ++m)
#pragma unroll
            for (int n = 0; n < 4; ++n)
                acc[m][n] = wmma_bf16(ah[m].v, bl[n].v, acc[m][n]);
    }

    // C/D layout: VGPR j -> row half*8+j (within 16-tile), col = lane%16
    __shared__ float S[KPAD * 65];         // stride 65: conflict-free column reads
    __shared__ float rowRes[KDIM];
    __shared__ float colRes[LDIM];
#pragma unroll
    for (int m = 0; m < 3; ++m)
#pragma unroll
        for (int n = 0; n < 4; ++n)
#pragma unroll
            for (int j = 0; j < 8; ++j)
                S[(m * 16 + half * 8 + j) * 65 + n * 16 + r] = acc[m][n][j];
    __syncthreads();   // single-wave: S_NOP + dscnt flush

    const int Ki = img_lens[i];
    const int Lt = cap_lens[t];

    // v2t: per valid image region k, softmax over valid words l, weighted sum.
    // Masked (NEG) entries contribute exp(~-2e4)==0, identical to reference.
    for (int k = lane; k < Ki; k += 32) {
        const float* rowp = &S[k * 65];
        float mx = -1e30f;
        for (int l = 0; l < Lt; ++l) mx = fmaxf(mx, rowp[l]);
        float den = 0.f, num = 0.f;
        for (int l = 0; l < Lt; ++l) {
            const float s = rowp[l];
            const float p = __expf((s - mx) * LAMB_INV);
            den += p; num += p * s;
        }
        rowRes[k] = num / den;
    }
    // t2v: per valid word l, softmax over valid regions k.
    for (int l = lane; l < Lt; l += 32) {
        float mx = -1e30f;
        for (int k = 0; k < Ki; ++k) mx = fmaxf(mx, S[k * 65 + l]);
        float den = 0.f, num = 0.f;
        for (int k = 0; k < Ki; ++k) {
            const float s = S[k * 65 + l];
            const float p = __expf((s - mx) * LAMB_INV);
            den += p; num += p * s;
        }
        colRes[l] = num / den;
    }
    __syncthreads();

    if (lane == 0) {
        float sr = 0.f;
        for (int k = 0; k < Ki; ++k) sr += rowRes[k];
        float sc = 0.f;
        for (int l = 0; l < Lt; ++l) sc += colRes[l];
        out[(size_t)i * BT + t] = 0.5f * (sr / (float)Ki + sc / (float)Lt);
    }
}

// ---------------------------------------------------------------------------
// Launcher. Inputs: [0]=img_cls(unused) [1]=imgs [2]=cap_cls(unused) [3]=caps
//                   [4]=img_lens(i32)   [5]=cap_lens(i32)
// Workspace: Ah|Al (128*48*1024 bf16 each) + Bh|Bl (128*64*1024 bf16 each)
// ---------------------------------------------------------------------------
extern "C" void kernel_launch(void* const* d_in, const int* in_sizes, int n_in,
                              void* d_out, int out_size, void* d_ws, size_t ws_size,
                              hipStream_t stream) {
    const float* imgs     = (const float*)d_in[1];
    const float* caps     = (const float*)d_in[3];
    const int*   img_lens = (const int*)d_in[4];
    const int*   cap_lens = (const int*)d_in[5];

    unsigned short* Ah = (unsigned short*)d_ws;
    unsigned short* Al = Ah + (size_t)BI * KPAD * DDIM;
    unsigned short* Bh = Al + (size_t)BI * KPAD * DDIM;
    unsigned short* Bl = Bh + (size_t)BT * LDIM * DDIM;

    const int nrows = BI * KPAD + BT * LDIM;   // 6144 + 8192
    normalize_split<<<dim3(nrows), dim3(128), 0, stream>>>(imgs, caps, Ah, Al, Bh, Bl);

    cam_block<<<dim3(BT, BI), dim3(32), 0, stream>>>(Ah, Al, Bh, Bl,
                                                     img_lens, cap_lens,
                                                     (float*)d_out);
}
